// PyTorchQwen2DecoderLayer_65936337928787
// MI455X (gfx1250) — matrix operand order repaired and checked by reading, not compiled, not run on hardware
//
#include <hip/hip_runtime.h>
#include <hip/hip_bf16.h>
#include <math.h>
#include <stdint.h>

// ---------------------------------------------------------------------------
// Qwen2 decoder layer for MI455X (gfx1250, wave32, WMMA).
// GEMMs + attention matmuls via v_wmma_f32_16x16x32_bf16 (f32 accumulate).
// GEMM tiles staged global->LDS with GLOBAL_LOAD_ASYNC_TO_LDS_B128 (ASYNCcnt).
// ---------------------------------------------------------------------------

typedef __bf16 bf16;
typedef __bf16 v8bf  __attribute__((ext_vector_type(8)));
typedef __bf16 v16bf __attribute__((ext_vector_type(16)));
typedef float  v8f   __attribute__((ext_vector_type(8)));

#define B_    2
#define S_    2048
#define H_    896
#define NH_   7
#define D_    128
#define I_    4864
#define M_TOK (B_ * S_)           // 4096 token rows
#define QKV_N (NH_ * D_ + 2 * D_) // 1152 (q 896 | k 128 | v 128)
#define GU_N  (2 * I_)            // 9728 (gate | up)

// toggle: 1 = stage GEMM tiles with async-to-LDS instructions, 0 = b128 copies
#define GEMM_ASYNC_LDS 1

__device__ __forceinline__ v16bf cat16(v8bf lo, v8bf hi) {
  return __builtin_shufflevector(lo, hi, 0,1,2,3,4,5,6,7,8,9,10,11,12,13,14,15);
}

// --------------------------- fp32 -> bf16 copy ------------------------------
__global__ void cvt_f32_bf16(const float* __restrict__ src, bf16* __restrict__ dst, long n) {
  long i = (long)blockIdx.x * blockDim.x + threadIdx.x;
  long stride = (long)gridDim.x * blockDim.x;
  for (; i < n; i += stride) dst[i] = (bf16)src[i];
}

// --------------------------- bias packing -----------------------------------
__global__ void pack_bias(const float* __restrict__ bq, const float* __restrict__ bk,
                          const float* __restrict__ bv, float* __restrict__ out) {
  int i = blockIdx.x * blockDim.x + threadIdx.x;
  if (i < NH_ * D_)            out[i] = bq[i];
  else if (i < NH_ * D_ + D_)  out[i] = bk[i - NH_ * D_];
  else if (i < QKV_N)          out[i] = bv[i - NH_ * D_ - D_];
}

// --------------------------- RMSNorm -> bf16 --------------------------------
__global__ __launch_bounds__(256) void rmsnorm_bf16(const float* __restrict__ x,
                                                    const float* __restrict__ w,
                                                    bf16* __restrict__ out) {
  __shared__ float red[256];
  const int row = blockIdx.x;
  const float* xr = x + (size_t)row * H_;
  float s = 0.f;
  for (int i = threadIdx.x; i < H_; i += 256) { float v = xr[i]; s += v * v; }
  red[threadIdx.x] = s;
  __syncthreads();
  for (int off = 128; off > 0; off >>= 1) {
    if (threadIdx.x < off) red[threadIdx.x] += red[threadIdx.x + off];
    __syncthreads();
  }
  const float scale = rsqrtf(red[0] / (float)H_ + 1e-6f);
  for (int i = threadIdx.x; i < H_; i += 256)
    out[(size_t)row * H_ + i] = (bf16)(xr[i] * scale * w[i]);
}

// --------------------------- bf16 WMMA GEMM ---------------------------------
// C[M,N] = A[M,K] @ W[N,K]^T (+bias[N]) (+resid[M,N]).
// Block: 256 thr = 8 waves; block tile 128x128, BK=32.
// Wave grid 4(M) x 2(N); wave tile 32x64 = 2x4 WMMA accs (8 WMMA / k-step).
template <bool HAS_BIAS, bool HAS_RES, bool OUT_BF16>
__global__ __launch_bounds__(256) void gemm_bf16(const bf16* __restrict__ A,
                                                 const bf16* __restrict__ W,
                                                 const float* __restrict__ bias,
                                                 const float* __restrict__ resid,
                                                 void* __restrict__ Cout,
                                                 int M, int N, int K) {
  __shared__ __align__(16) bf16 As[128][48]; // 96B rows: 16B-aligned, padded
  __shared__ __align__(16) bf16 Bs[128][48];
  const int tid  = threadIdx.x;
  const int lane = tid & 31, wid = tid >> 5;
  const int lh   = lane & 15, sel = lane >> 4;   // lane half select
  const int m_base = (wid & 3) * 32, n_base = (wid >> 2) * 64;
  const int bm = blockIdx.y * 128, bn = blockIdx.x * 128;

  v8f acc[2][4];
#pragma unroll
  for (int i = 0; i < 2; i++)
#pragma unroll
    for (int j = 0; j < 4; j++)
#pragma unroll
      for (int e = 0; e < 8; e++) acc[i][j][e] = 0.f;

  for (int kk = 0; kk < K; kk += 32) {
    // ---- stage A/B tiles (128 rows x 32 bf16 each = 512 b128 chunks each) --
#if GEMM_ASYNC_LDS
#pragma unroll
    for (int t = 0; t < 2; t++) {
      int idx = tid + t * 256;
      int row = idx >> 2, c4 = idx & 3;
      const bf16* ga = A + (size_t)(bm + row) * K + kk + c4 * 8;
      const bf16* gb = W + (size_t)(bn + row) * K + kk + c4 * 8;
      uint32_t la = (uint32_t)(uintptr_t)&As[row][c4 * 8]; // generic addr[31:0] = LDS offset
      uint32_t lb = (uint32_t)(uintptr_t)&Bs[row][c4 * 8];
      asm volatile("global_load_async_to_lds_b128 %0, %1, off"
                   :: "v"(la), "v"(ga) : "memory");
      asm volatile("global_load_async_to_lds_b128 %0, %1, off"
                   :: "v"(lb), "v"(gb) : "memory");
    }
    asm volatile("s_wait_asynccnt 0" ::: "memory");
#else
#pragma unroll
    for (int t = 0; t < 2; t++) {
      int idx = tid + t * 256;
      int row = idx >> 2, c4 = idx & 3;
      const float4* ga = reinterpret_cast<const float4*>(A + (size_t)(bm + row) * K + kk + c4 * 8);
      const float4* gb = reinterpret_cast<const float4*>(W + (size_t)(bn + row) * K + kk + c4 * 8);
      *reinterpret_cast<float4*>(&As[row][c4 * 8]) = *ga;
      *reinterpret_cast<float4*>(&Bs[row][c4 * 8]) = *gb;
    }
#endif
    __syncthreads();

    // prefetch next K-step tiles into cache (global_prefetch_b8)
    if (kk + 32 < K) {
      int row = tid >> 1;
      if (tid & 1) __builtin_prefetch(W + (size_t)(bn + row) * K + kk + 32, 0, 1);
      else         __builtin_prefetch(A + (size_t)(bm + row) * K + kk + 32, 0, 1);
    }

    // ---- fragments + 8 WMMA -------------------------------------------------
    v16bf afr[2], bfr[4];
#pragma unroll
    for (int mt = 0; mt < 2; mt++) {
      const bf16* ap = &As[m_base + mt * 16 + lh][sel * 8];
      afr[mt] = cat16(*reinterpret_cast<const v8bf*>(ap),
                      *reinterpret_cast<const v8bf*>(ap + 16));
    }
#pragma unroll
    for (int nt = 0; nt < 4; nt++) {
      const bf16* bp = &Bs[n_base + nt * 16 + lh][sel * 8];
      bfr[nt] = cat16(*reinterpret_cast<const v8bf*>(bp),
                      *reinterpret_cast<const v8bf*>(bp + 16));
    }
#pragma unroll
    for (int mt = 0; mt < 2; mt++)
#pragma unroll
      for (int nt = 0; nt < 4; nt++)
        acc[mt][nt] = __builtin_amdgcn_wmma_f32_16x16x32_bf16(
            false, afr[mt], false, bfr[nt], (short)0, acc[mt][nt], false, false);
    __syncthreads();
  }

  // epilogue (branch-free per template): C layout — lane=col, VGPR r=row
#pragma unroll
  for (int mt = 0; mt < 2; mt++)
#pragma unroll
    for (int nt = 0; nt < 4; nt++)
#pragma unroll
      for (int r = 0; r < 8; r++) {
        int m = bm + m_base + mt * 16 + (sel ? r + 8 : r);
        int n = bn + n_base + nt * 16 + lh;
        float v = acc[mt][nt][r];
        if (HAS_BIAS) v += bias[n];
        if (HAS_RES)  v += resid[(size_t)m * N + n];
        if (OUT_BF16) ((bf16*)Cout)[(size_t)m * N + n] = (bf16)v;
        else          ((float*)Cout)[(size_t)m * N + n] = v;
      }
}

// --------------------------- RoPE + reformat --------------------------------
// qkv fp32 [B*S,1152] -> q bf16 [B,NH,S,D] (rope), k bf16 [B,S,D] (rope),
// vT bf16 [B,D,S] (transposed so PV's B-operand tiles are contiguous rows).
__global__ void rope_reformat(const float* __restrict__ qkv,
                              const float* __restrict__ cosT, const float* __restrict__ sinT,
                              bf16* __restrict__ qo, bf16* __restrict__ ko, bf16* __restrict__ vto) {
  const int bs = blockIdx.x;
  const int b = bs / S_, s = bs % S_;
  const float* row = qkv + (size_t)bs * QKV_N;
  const float* cr = cosT + (size_t)s * D_;
  const float* sr = sinT + (size_t)s * D_;
  for (int idx = threadIdx.x; idx < NH_ * 64; idx += blockDim.x) {
    int h = idx >> 6, d = idx & 63;
    float x1 = row[h * D_ + d], x2 = row[h * D_ + d + 64];
    float c = cr[d], sn = sr[d];
    size_t o = ((size_t)(b * NH_ + h) * S_ + s) * D_;
    qo[o + d]      = (bf16)(x1 * c - x2 * sn);
    qo[o + d + 64] = (bf16)(x1 * sn + x2 * c);
  }
  for (int d = threadIdx.x; d < 64; d += blockDim.x) {
    float x1 = row[NH_ * D_ + d], x2 = row[NH_ * D_ + d + 64];
    float c = cr[d], sn = sr[d];
    size_t o = ((size_t)b * S_ + s) * D_;
    ko[o + d]      = (bf16)(x1 * c - x2 * sn);
    ko[o + d + 64] = (bf16)(x1 * sn + x2 * c);
  }
  for (int d = threadIdx.x; d < D_; d += blockDim.x)
    vto[((size_t)(b * D_ + d)) * S_ + s] = (bf16)row[NH_ * D_ + D_ + d];
}

// --------------------------- flash attention --------------------------------
// grid (S/64, B*NH); 128 thr = 4 waves; each wave owns 16 q rows.
// Per 64-key tile: 16 WMMA (QK^T, K-dim=128) + 16 WMMA (PV, K-dim=64).
__global__ __launch_bounds__(128) void flash_attn(const bf16* __restrict__ Q,
                                                  const bf16* __restrict__ Kc,
                                                  const bf16* __restrict__ Vt,
                                                  bf16* __restrict__ Oa) {
  __shared__ __align__(16) bf16 Ks[64][136];   // K tile  [key][d], padded
  __shared__ __align__(16) bf16 Vts[128][72];  // V^T tile [d][key], padded
  __shared__ __align__(16) bf16 Pl[4][16][72]; // per-wave P staging (C->A relayout)
  const int tid = threadIdx.x, lane = tid & 31, w = tid >> 5;
  const int lh = lane & 15, sel = lane >> 4;
  const int q0 = blockIdx.x * 64;
  const int bh = blockIdx.y;
  const int b = bh / NH_, h = bh % NH_;
  const float scale = 0.08838834764831843f; // D^-0.5

  // Q A-fragments (16 rows x 128 K) loaded once: 4 k-steps of 32
  v16bf qa[4];
  {
    const bf16* qptr = Q + ((size_t)bh * S_ + q0 + w * 16 + lh) * D_;
#pragma unroll
    for (int ks = 0; ks < 4; ks++)
      qa[ks] = cat16(*reinterpret_cast<const v8bf*>(qptr + ks * 32 + sel * 8),
                     *reinterpret_cast<const v8bf*>(qptr + ks * 32 + sel * 8 + 16));
  }

  v8f o[8];
#pragma unroll
  for (int dt = 0; dt < 8; dt++)
#pragma unroll
    for (int e = 0; e < 8; e++) o[dt][e] = 0.f;
  float m_row[8], l_row[8];
#pragma unroll
  for (int r = 0; r < 8; r++) { m_row[r] = -1e30f; l_row[r] = 0.f; }

  const int nkt = blockIdx.x + 1; // causal: key tiles 0..q-tile
  for (int kt = 0; kt < nkt; kt++) {
    // stage K (64x128) and V^T (128x64) tiles, b128 copies
#pragma unroll
    for (int i = 0; i < 8; i++) {
      int idx = tid + i * 128;
      { int row = idx >> 4, c4 = idx & 15;
        const float4* gp = reinterpret_cast<const float4*>(
            Kc + ((size_t)b * S_ + kt * 64 + row) * D_ + c4 * 8);
        *reinterpret_cast<float4*>(&Ks[row][c4 * 8]) = *gp; }
      { int row = idx >> 3, c4 = idx & 7;
        const float4* gp = reinterpret_cast<const float4*>(
            Vt + ((size_t)(b * D_ + row)) * S_ + kt * 64 + c4 * 8);
        *reinterpret_cast<float4*>(&Vts[row][c4 * 8]) = *gp; }
    }
    __syncthreads();
    if (kt + 1 < nkt) { // prefetch next key tile
      __builtin_prefetch(Kc + ((size_t)b * S_ + (kt + 1) * 64 + (tid >> 1)) * D_, 0, 1);
      __builtin_prefetch(Vt + ((size_t)(b * D_ + tid)) * S_ + (kt + 1) * 64, 0, 1);
    }

    // scores: 16x64 in 4 N-subtiles, K-dim 128 in 4 steps
    v8f sc[4];
#pragma unroll
    for (int nt = 0; nt < 4; nt++)
#pragma unroll
      for (int e = 0; e < 8; e++) sc[nt][e] = 0.f;
#pragma unroll
    for (int ks = 0; ks < 4; ks++)
#pragma unroll
      for (int nt = 0; nt < 4; nt++) {
        const bf16* bp = &Ks[nt * 16 + lh][ks * 32 + sel * 8];
        v16bf bfr = cat16(*reinterpret_cast<const v8bf*>(bp),
                          *reinterpret_cast<const v8bf*>(bp + 16));
        sc[nt] = __builtin_amdgcn_wmma_f32_16x16x32_bf16(
            false, qa[ks], false, bfr, (short)0, sc[nt], false, false);
      }

    // causal mask + online softmax (row = per-r per-lane-half)
    float p[4][8], rm[8];
#pragma unroll
    for (int r = 0; r < 8; r++) rm[r] = -1e30f;
#pragma unroll
    for (int nt = 0; nt < 4; nt++)
#pragma unroll
      for (int r = 0; r < 8; r++) {
        int col  = kt * 64 + nt * 16 + lh;
        int rowg = q0 + w * 16 + (sel ? r + 8 : r);
        float v = sc[nt][r] * scale;
        if (col > rowg) v = -1e30f;
        p[nt][r] = v;
        rm[r] = fmaxf(rm[r], v);
      }
    for (int mk = 1; mk < 16; mk <<= 1)
#pragma unroll
      for (int r = 0; r < 8; r++) rm[r] = fmaxf(rm[r], __shfl_xor(rm[r], mk, 32));
    float corr[8], rs[8];
#pragma unroll
    for (int r = 0; r < 8; r++) {
      float mn = fmaxf(m_row[r], rm[r]);
      corr[r] = __expf(m_row[r] - mn);
      m_row[r] = mn;
      rs[r] = 0.f;
    }
#pragma unroll
    for (int nt = 0; nt < 4; nt++)
#pragma unroll
      for (int r = 0; r < 8; r++) {
        float e = __expf(p[nt][r] - m_row[r]);
        p[nt][r] = e;
        rs[r] += e;
      }
    for (int mk = 1; mk < 16; mk <<= 1)
#pragma unroll
      for (int r = 0; r < 8; r++) rs[r] += __shfl_xor(rs[r], mk, 32);
#pragma unroll
    for (int r = 0; r < 8; r++) l_row[r] = l_row[r] * corr[r] + rs[r];
#pragma unroll
    for (int dt = 0; dt < 8; dt++)
#pragma unroll
      for (int r = 0; r < 8; r++) o[dt][r] *= corr[r];

    // relayout P (C layout) -> A layout via per-wave LDS staging
#pragma unroll
    for (int nt = 0; nt < 4; nt++)
#pragma unroll
      for (int r = 0; r < 8; r++)
        Pl[w][sel ? r + 8 : r][nt * 16 + lh] = (bf16)p[nt][r];
    asm volatile("s_wait_dscnt 0" ::: "memory"); // per-wave LDS RAW fence

    // PV: O(16x128) += P(16x64) @ V^T-as-B, K-dim 64 in 2 steps
#pragma unroll
    for (int kp = 0; kp < 2; kp++) {
      const bf16* ap = &Pl[w][lh][kp * 32 + sel * 8];
      v16bf afr = cat16(*reinterpret_cast<const v8bf*>(ap),
                        *reinterpret_cast<const v8bf*>(ap + 16));
#pragma unroll
      for (int dt = 0; dt < 8; dt++) {
        const bf16* bp = &Vts[dt * 16 + lh][kp * 32 + sel * 8];
        v16bf bfr = cat16(*reinterpret_cast<const v8bf*>(bp),
                          *reinterpret_cast<const v8bf*>(bp + 16));
        o[dt] = __builtin_amdgcn_wmma_f32_16x16x32_bf16(
            false, afr, false, bfr, (short)0, o[dt], false, false);
      }
    }
    __syncthreads();
  }

  // normalize and write bf16 attn output in [B,S,NH*D] layout
#pragma unroll
  for (int dt = 0; dt < 8; dt++)
#pragma unroll
    for (int r = 0; r < 8; r++) {
      int rowg = q0 + w * 16 + (sel ? r + 8 : r);
      float v = o[dt][r] / l_row[r];
      Oa[((size_t)(b * S_ + rowg)) * H_ + h * D_ + dt * 16 + lh] = (bf16)v;
    }
}

// --------------------------- SwiGLU -----------------------------------------
__global__ void swiglu(const bf16* __restrict__ gu, bf16* __restrict__ act, long n) {
  long i = (long)blockIdx.x * blockDim.x + threadIdx.x;
  long stride = (long)gridDim.x * blockDim.x;
  for (; i < n; i += stride) {
    long m = i / I_, j = i % I_;
    float g = (float)gu[m * GU_N + j];
    float u = (float)gu[m * GU_N + I_ + j];
    act[i] = (bf16)((g / (1.f + __expf(-g))) * u);
  }
}

// --------------------------- launcher ----------------------------------------
extern "C" void kernel_launch(void* const* d_in, const int* in_sizes, int n_in,
                              void* d_out, int out_size, void* d_ws, size_t ws_size,
                              hipStream_t stream) {
  (void)in_sizes; (void)n_in; (void)out_size; (void)ws_size;
  const float* hidden = (const float*)d_in[0];
  const float* wq  = (const float*)d_in[1];
  const float* bq  = (const float*)d_in[2];
  const float* wk  = (const float*)d_in[3];
  const float* bk  = (const float*)d_in[4];
  const float* wv  = (const float*)d_in[5];
  const float* bv  = (const float*)d_in[6];
  const float* wo  = (const float*)d_in[7];
  const float* wg  = (const float*)d_in[8];
  const float* wu  = (const float*)d_in[9];
  const float* wd  = (const float*)d_in[10];
  const float* ln1 = (const float*)d_in[11];
  const float* ln2 = (const float*)d_in[12];
  // d_in[13] = attention_mask (causal recomputed in-kernel)
  const float* cosT = (const float*)d_in[14];
  const float* sinT = (const float*)d_in[15];

  char* p = (char*)d_ws;
  auto take = [&](size_t bytes) -> char* {
    char* r = p;
    p += (bytes + 255) & ~(size_t)255;
    return r;
  };
  bf16*  wqkv_b = (bf16*) take((size_t)QKV_N * H_ * 2);
  bf16*  wo_b   = (bf16*) take((size_t)H_ * H_ * 2);
  bf16*  wgu_b  = (bf16*) take((size_t)GU_N * H_ * 2);
  bf16*  wdn_b  = (bf16*) take((size_t)H_ * I_ * 2);
  float* bqkv   = (float*)take((size_t)QKV_N * 4);
  bf16*  h_b    = (bf16*) take((size_t)M_TOK * H_ * 2);
  float* qkv_f  = (float*)take((size_t)M_TOK * QKV_N * 4);
  bf16*  q_b    = (bf16*) take((size_t)B_ * NH_ * S_ * D_ * 2);
  bf16*  k_b    = (bf16*) take((size_t)B_ * S_ * D_ * 2);
  bf16*  vt_b   = (bf16*) take((size_t)B_ * D_ * S_ * 2);
  bf16*  attn_b = (bf16*) take((size_t)M_TOK * H_ * 2);
  float* x1_f   = (float*)take((size_t)M_TOK * H_ * 4);
  bf16*  h2_b   = (bf16*) take((size_t)M_TOK * H_ * 2);
  bf16*  gu_b   = (bf16*) take((size_t)M_TOK * GU_N * 2);
  bf16*  act_b  = (bf16*) take((size_t)M_TOK * I_ * 2);

  auto cvt = [&](const float* s, bf16* d, long n) {
    long blocks = (n + 255) / 256; if (blocks > 8192) blocks = 8192;
    cvt_f32_bf16<<<dim3((unsigned)blocks), dim3(256), 0, stream>>>(s, d, n);
  };

  // weights -> bf16 (L2-resident working set)
  cvt(wq, wqkv_b,                        (long)NH_ * D_ * H_);
  cvt(wk, wqkv_b + (long)NH_ * D_ * H_,  (long)D_ * H_);
  cvt(wv, wqkv_b + (long)(NH_ * D_ + D_) * H_, (long)D_ * H_);
  cvt(wo, wo_b,                          (long)H_ * H_);
  cvt(wg, wgu_b,                         (long)I_ * H_);
  cvt(wu, wgu_b + (long)I_ * H_,         (long)I_ * H_);
  cvt(wd, wdn_b,                         (long)H_ * I_);
  pack_bias<<<dim3((QKV_N + 255) / 256), dim3(256), 0, stream>>>(bq, bk, bv, bqkv);

  // attention half
  rmsnorm_bf16<<<dim3(M_TOK), dim3(256), 0, stream>>>(hidden, ln1, h_b);
  gemm_bf16<true, false, false><<<dim3(QKV_N / 128, M_TOK / 128), dim3(256), 0, stream>>>(
      h_b, wqkv_b, bqkv, nullptr, qkv_f, M_TOK, QKV_N, H_);
  rope_reformat<<<dim3(M_TOK), dim3(128), 0, stream>>>(qkv_f, cosT, sinT, q_b, k_b, vt_b);
  flash_attn<<<dim3(S_ / 64, B_ * NH_), dim3(128), 0, stream>>>(q_b, k_b, vt_b, attn_b);
  gemm_bf16<false, true, false><<<dim3(H_ / 128, M_TOK / 128), dim3(256), 0, stream>>>(
      attn_b, wo_b, nullptr, hidden, x1_f, M_TOK, H_, H_);

  // MLP half
  rmsnorm_bf16<<<dim3(M_TOK), dim3(256), 0, stream>>>(x1_f, ln2, h2_b);
  gemm_bf16<false, false, true><<<dim3(GU_N / 128, M_TOK / 128), dim3(256), 0, stream>>>(
      h2_b, wgu_b, nullptr, nullptr, gu_b, M_TOK, GU_N, H_);
  swiglu<<<dim3(8192), dim3(256), 0, stream>>>(gu_b, act_b, (long)M_TOK * I_);
  gemm_bf16<false, true, false><<<dim3(H_ / 128, M_TOK / 128), dim3(256), 0, stream>>>(
      act_b, wdn_b, nullptr, x1_f, d_out, M_TOK, H_, I_);
}
